// Swin_70755291234892
// MI455X (gfx1250) — compile-verified, matrix operand order checked by reading, not run on hardware
//
#include <hip/hip_runtime.h>

typedef __attribute__((ext_vector_type(16))) __bf16 v16bf;
typedef __attribute__((ext_vector_type(8)))  float  v8f;
typedef __attribute__((ext_vector_type(4)))  unsigned int u32x4;

#define TOKENS 200704   // 64 * 56 * 56
#define SDIM 96
#define SHEADS 3
#define SHD 32
#define SWS 7
#define SL 49
#define HF 56
#define SSHIFT 3

__device__ __forceinline__ unsigned short f2bf(float f) {
    union { float f; unsigned u; } v; v.f = f;
    unsigned u = v.u;
    u += 0x7FFFu + ((u >> 16) & 1u);   // round-to-nearest-even
    return (unsigned short)(u >> 16);
}

// ---------------------------------------------------------------- TDM: 2D tile global -> LDS
// rows x cols (elements, 2B each), contiguous rows (stride = cols), contiguous LDS dest.
__device__ __forceinline__ void tdm_load_2d_bf16(const unsigned short* gsrc,
                                                 unsigned lds_byte_addr,
                                                 unsigned cols, unsigned rows) {
    unsigned long ga = (unsigned long)gsrc;
    unsigned g0w0 = 1u;                                   // count=1, is_restore=0
    unsigned g0w1 = lds_byte_addr;                        // lds_addr
    unsigned g0w2 = (unsigned)ga;                         // global_addr[31:0]
    unsigned g0w3 = (unsigned)((ga >> 32) & 0x1FFFFFFu)   // global_addr[56:32]
                    | 0x80000000u;                        // type=2 (bits 127:126)
    unsigned g1w0 = 0x00010000u;                          // data_size=1 (2 bytes)
    unsigned g1w1 = (cols & 0xFFFFu) << 16;               // tensor_dim0[15:0]
    unsigned g1w2 = (cols >> 16) | ((rows & 0xFFFFu) << 16); // dim0[31:16] | dim1[15:0]
    unsigned g1w3 = (rows >> 16) | (cols << 16);          // dim1[31:16] | tile_dim0
    unsigned g1w4 = rows;                                 // tile_dim1 | tile_dim2=0
    unsigned g1w5 = cols;                                 // tensor_dim0_stride[31:0]
    unsigned g1w6 = 0u;                                   // stride0[47:32] | stride1[15:0]
    unsigned g1w7 = 0u;                                   // stride1[47:16]
    asm volatile(
        "s_mov_b32 s80, %0\n\t"
        "s_mov_b32 s81, %1\n\t"
        "s_mov_b32 s82, %2\n\t"
        "s_mov_b32 s83, %3\n\t"
        "s_mov_b32 s84, %4\n\t"
        "s_mov_b32 s85, %5\n\t"
        "s_mov_b32 s86, %6\n\t"
        "s_mov_b32 s87, %7\n\t"
        "s_mov_b32 s88, %8\n\t"
        "s_mov_b32 s89, %9\n\t"
        "s_mov_b32 s90, %10\n\t"
        "s_mov_b32 s91, %11\n\t"
        "tensor_load_to_lds s[80:83], s[84:91]\n\t"
        :
        : "s"(g0w0), "s"(g0w1), "s"(g0w2), "s"(g0w3),
          "s"(g1w0), "s"(g1w1), "s"(g1w2), "s"(g1w3),
          "s"(g1w4), "s"(g1w5), "s"(g1w6), "s"(g1w7)
        : "s80","s81","s82","s83","s84","s85","s86","s87","s88","s89","s90","s91",
          "memory");
    __builtin_amdgcn_s_wait_tensorcnt(0);
}

// ---------------------------------------------------------------- weight cast
__global__ void cast_f32_to_bf16(const float* __restrict__ src,
                                 unsigned short* __restrict__ dst, int n) {
    int i = blockIdx.x * blockDim.x + threadIdx.x;
    if (i < n) dst[i] = f2bf(src[i]);
}

// ---------------------------------------------------------------- patch embed
__global__ __launch_bounds__(256) void patch_embed(const float* __restrict__ img,
                                                   const float* __restrict__ w,
                                                   const float* __restrict__ b,
                                                   float* __restrict__ out) {
    long i = (long)blockIdx.x * 256 + threadIdx.x;
    if (i >= (long)TOKENS * SDIM) return;
    int c = (int)(i % SDIM);
    long tok = i / SDIM;
    int bidx = (int)(tok / 3136);
    int hw = (int)(tok % 3136);
    int ph = hw / HF, pw = hw % HF;
    float acc = b[c];
    const float* wc = w + c * 48;
#pragma unroll
    for (int ci = 0; ci < 3; ci++) {
        const float* ip = img + (((long)bidx * 3 + ci) * 224 + ph * 4) * 224 + pw * 4;
#pragma unroll
        for (int kh = 0; kh < 4; kh++)
#pragma unroll
            for (int kw = 0; kw < 4; kw++)
                acc += ip[kh * 224 + kw] * wc[ci * 16 + kh * 4 + kw];
    }
    out[tok * SDIM + c] = acc;
}

// ---------------------------------------------------------------- layernorm
template <int OUTBF>
__global__ __launch_bounds__(256) void layernorm_k(const float* __restrict__ X,
                                                   const float* __restrict__ g,
                                                   const float* __restrict__ bta,
                                                   float* __restrict__ outF,
                                                   unsigned short* __restrict__ outB) {
    int wave = threadIdx.x >> 5, lane = threadIdx.x & 31;
    long tok = (long)blockIdx.x * 8 + wave;
    if (tok >= TOKENS) return;
    const float* x = X + tok * SDIM;
    float v0 = x[lane], v1 = x[lane + 32], v2 = x[lane + 64];
    float s = v0 + v1 + v2;
    for (int o = 16; o; o >>= 1) s += __shfl_xor(s, o, 32);
    float mean = s * (1.0f / 96.0f);
    float d0 = v0 - mean, d1 = v1 - mean, d2 = v2 - mean;
    float q = d0 * d0 + d1 * d1 + d2 * d2;
    for (int o = 16; o; o >>= 1) q += __shfl_xor(q, o, 32);
    float rstd = rsqrtf(q * (1.0f / 96.0f) + 1e-5f);
    float r0 = d0 * rstd * g[lane]      + bta[lane];
    float r1 = d1 * rstd * g[lane + 32] + bta[lane + 32];
    float r2 = d2 * rstd * g[lane + 64] + bta[lane + 64];
    if (OUTBF) {
        outB[tok * SDIM + lane]      = f2bf(r0);
        outB[tok * SDIM + lane + 32] = f2bf(r1);
        outB[tok * SDIM + lane + 64] = f2bf(r2);
    } else {
        outF[tok * SDIM + lane]      = r0;
        outF[tok * SDIM + lane + 32] = r1;
        outF[tok * SDIM + lane + 64] = r2;
    }
}

// ---------------------------------------------------------------- bf16 WMMA GEMM, LDS-staged W
// C[M, NT*16] strip per block: W slice (NT*16 rows x K) staged into LDS by the TDM,
// each wave owns 16 rows x NT*16 cols; A fragment loaded once per K-step, reused NT times.
// EPI 0: bias->bf16 | 1: bias+GELU->bf16 | 2: bias+residual->f32
template <int EPI, int K, int NT>
__global__ __launch_bounds__(128) void gemm_wmma(const unsigned short* __restrict__ A,
                                                 const unsigned short* __restrict__ W,
                                                 const float* __restrict__ bias,
                                                 const float* __restrict__ res,
                                                 float* __restrict__ outF,
                                                 unsigned short* __restrict__ outB,
                                                 int N) {
    __shared__ __attribute__((aligned(16))) unsigned short Wlds[NT * 16 * K];
    int tid = threadIdx.x;
    int wave = tid >> 5, lane = tid & 31, hlf = lane >> 4, l15 = lane & 15;
    int ct0 = blockIdx.x * (NT * 16);

    if (wave == 0) {
        tdm_load_2d_bf16(W + (long)ct0 * K,
                         (unsigned)(unsigned long)(uintptr_t)&Wlds[0],
                         (unsigned)K, (unsigned)(NT * 16));
    }
    __syncthreads();

    long rt = ((long)blockIdx.y * 4 + wave) * 16;
    const unsigned short* arow = A + (rt + l15) * K + hlf * 8;
    const unsigned short* wl = &Wlds[0] + l15 * K + hlf * 8;

    v8f acc[NT];
#pragma unroll
    for (int nt = 0; nt < NT; nt++) acc[nt] = (v8f){};

#pragma unroll
    for (int k0 = 0; k0 < K; k0 += 32) {
        v16bf af;
        u32x4* ap = (u32x4*)&af;
        ap[0] = *(const u32x4*)(arow + k0);
        ap[1] = *(const u32x4*)(arow + k0 + 16);
        if (k0 + 32 < K) __builtin_prefetch(arow + k0 + 32, 0, 0);
#pragma unroll
        for (int nt = 0; nt < NT; nt++) {
            v16bf bfm;
            u32x4* bp = (u32x4*)&bfm;
            bp[0] = *(const u32x4*)(wl + nt * 16 * K + k0);
            bp[1] = *(const u32x4*)(wl + nt * 16 * K + k0 + 16);
            acc[nt] = __builtin_amdgcn_wmma_f32_16x16x32_bf16(false, af, false, bfm,
                                                              (short)0, acc[nt], false, false);
        }
    }

#pragma unroll
    for (int nt = 0; nt < NT; nt++) {
        int col = ct0 + nt * 16 + l15;
        float bv = bias[col];
#pragma unroll
        for (int r = 0; r < 8; r++) {
            long row = rt + hlf * 8 + r;
            float v = acc[nt][r] + bv;
            if (EPI == 0) {
                outB[row * N + col] = f2bf(v);
            } else if (EPI == 1) {
                float gl = 0.5f * v * (1.0f + erff(v * 0.70710678118654752f));
                outB[row * N + col] = f2bf(gl);
            } else {
                outF[row * N + col] = v + res[row * N + col];
            }
        }
    }
}

// ---------------------------------------------------------------- window attention
// one workgroup = (window, head); 4 waves. L=49 padded to 64.
__global__ __launch_bounds__(128) void win_attn(const unsigned short* __restrict__ QKV,
                                                const float* __restrict__ rpb,
                                                unsigned short* __restrict__ Aout,
                                                int shift) {
    __shared__ __attribute__((aligned(16))) unsigned short Qs[64][32];
    __shared__ __attribute__((aligned(16))) unsigned short Ks[64][32];
    __shared__ __attribute__((aligned(16))) unsigned short Vt[32][64];  // transposed V
    __shared__ __attribute__((aligned(16))) float          Ss[64][64];
    __shared__ __attribute__((aligned(16))) unsigned short Ps[64][64];

    int win = blockIdx.x, head = blockIdx.y;
    int b = win >> 6, wh = (win >> 3) & 7, ww = win & 7;
    int tid = threadIdx.x;

    for (int m = tid; m < 192; m += 128) {
        int mat = m >> 6, row = m & 63;
        if (row < SL) {
            int gh = wh * SWS + row / SWS, gw = ww * SWS + row % SWS;
            int sh = gh + shift; if (sh >= HF) sh -= HF;
            int sw = gw + shift; if (sw >= HF) sw -= HF;
            long tok = (long)b * 3136 + sh * HF + sw;
            const unsigned short* src = QKV + tok * 288 + mat * 96 + head * SHD;
            if (mat == 0) {
                u32x4* d = (u32x4*)&Qs[row][0]; const u32x4* s = (const u32x4*)src;
                d[0] = s[0]; d[1] = s[1]; d[2] = s[2]; d[3] = s[3];
            } else if (mat == 1) {
                u32x4* d = (u32x4*)&Ks[row][0]; const u32x4* s = (const u32x4*)src;
                d[0] = s[0]; d[1] = s[1]; d[2] = s[2]; d[3] = s[3];
            } else {
                for (int d = 0; d < 32; d++) Vt[d][row] = src[d];
            }
        } else {
            if (mat == 0)      { for (int d = 0; d < 32; d++) Qs[row][d] = 0; }
            else if (mat == 1) { for (int d = 0; d < 32; d++) Ks[row][d] = 0; }
            else               { for (int d = 0; d < 32; d++) Vt[d][row] = 0; }
        }
    }
    __syncthreads();

    int wave = tid >> 5, lane = tid & 31, hlf = lane >> 4, l15 = lane & 15;
    int m0 = wave * 16;

    // S = Q @ K^T
    v16bf qf;
    { u32x4* p = (u32x4*)&qf;
      p[0] = *(u32x4*)&Qs[m0 + l15][hlf * 8];
      p[1] = *(u32x4*)&Qs[m0 + l15][hlf * 8 + 16]; }
#pragma unroll
    for (int j = 0; j < 4; j++) {
        v16bf kf;
        { u32x4* p = (u32x4*)&kf;
          p[0] = *(u32x4*)&Ks[j * 16 + l15][hlf * 8];
          p[1] = *(u32x4*)&Ks[j * 16 + l15][hlf * 8 + 16]; }
        v8f sacc = {};
        sacc = __builtin_amdgcn_wmma_f32_16x16x32_bf16(false, qf, false, kf,
                                                       (short)0, sacc, false, false);
#pragma unroll
        for (int r = 0; r < 8; r++)
            Ss[m0 + hlf * 8 + r][j * 16 + l15] = sacc[r];
    }
    __syncthreads();

    // softmax rows (scale + relative-position bias + shift mask)
    if (tid < 64) {
        int i = tid;
        if (i < SL) {
            int ih = i / SWS, iw = i % SWS;
            int labi = 0;
            if (shift) {
                int gh = wh * SWS + ih, gw = ww * SWS + iw;
                labi = (gh < HF - SWS ? 0 : (gh < HF - SSHIFT ? 1 : 2)) * 3 +
                       (gw < HF - SWS ? 0 : (gw < HF - SSHIFT ? 1 : 2));
            }
            float mx = -1e30f;
            for (int j = 0; j < SL; j++) {
                int jh = j / SWS, jw = j % SWS;
                float bias = rpb[((ih - jh + 6) * 13 + (iw - jw + 6)) * 3 + head];
                float v = Ss[i][j] * 0.17677669529663687f + bias;
                if (shift) {
                    int gjh = wh * SWS + jh, gjw = ww * SWS + jw;
                    int labj = (gjh < HF - SWS ? 0 : (gjh < HF - SSHIFT ? 1 : 2)) * 3 +
                               (gjw < HF - SWS ? 0 : (gjw < HF - SSHIFT ? 1 : 2));
                    if (labj != labi) v -= 100.0f;
                }
                Ss[i][j] = v;
                mx = fmaxf(mx, v);
            }
            float sum = 0.f;
            for (int j = 0; j < SL; j++) {
                float e = __expf(Ss[i][j] - mx);
                Ss[i][j] = e;
                sum += e;
            }
            float inv = 1.0f / sum;
            for (int j = 0; j < SL; j++) Ps[i][j] = f2bf(Ss[i][j] * inv);
            for (int j = SL; j < 64; j++) Ps[i][j] = 0;
        } else {
            for (int j = 0; j < 64; j++) Ps[i][j] = 0;
        }
    }
    __syncthreads();

    // O = P @ V
    v8f oacc0 = {}, oacc1 = {};
#pragma unroll
    for (int ks = 0; ks < 2; ks++) {
        v16bf pf;
        { u32x4* p = (u32x4*)&pf;
          p[0] = *(u32x4*)&Ps[m0 + l15][ks * 32 + hlf * 8];
          p[1] = *(u32x4*)&Ps[m0 + l15][ks * 32 + hlf * 8 + 16]; }
        v16bf v0f, v1f;
        { u32x4* p = (u32x4*)&v0f;
          p[0] = *(u32x4*)&Vt[l15][ks * 32 + hlf * 8];
          p[1] = *(u32x4*)&Vt[l15][ks * 32 + hlf * 8 + 16]; }
        { u32x4* p = (u32x4*)&v1f;
          p[0] = *(u32x4*)&Vt[16 + l15][ks * 32 + hlf * 8];
          p[1] = *(u32x4*)&Vt[16 + l15][ks * 32 + hlf * 8 + 16]; }
        oacc0 = __builtin_amdgcn_wmma_f32_16x16x32_bf16(false, pf, false, v0f,
                                                        (short)0, oacc0, false, false);
        oacc1 = __builtin_amdgcn_wmma_f32_16x16x32_bf16(false, pf, false, v1f,
                                                        (short)0, oacc1, false, false);
    }
#pragma unroll
    for (int r = 0; r < 8; r++) {
        int row = m0 + hlf * 8 + r;
        if (row < SL) {
            int gh = wh * SWS + row / SWS, gw = ww * SWS + row % SWS;
            int sh = gh + shift; if (sh >= HF) sh -= HF;
            int sw = gw + shift; if (sw >= HF) sw -= HF;
            long tok = (long)b * 3136 + sh * HF + sw;
            Aout[tok * SDIM + head * SHD + l15]      = f2bf(oacc0[r]);
            Aout[tok * SDIM + head * SHD + 16 + l15] = f2bf(oacc1[r]);
        }
    }
}

// ---------------------------------------------------------------- NHWC -> NCHW
__global__ __launch_bounds__(256) void to_nchw(const float* __restrict__ X,
                                               float* __restrict__ out) {
    long i = (long)blockIdx.x * 256 + threadIdx.x;
    if (i >= (long)TOKENS * SDIM) return;
    int c = (int)(i % SDIM);
    long tok = i / SDIM;
    int b = (int)(tok / 3136);
    int hw = (int)(tok % 3136);
    out[((long)b * SDIM + c) * 3136 + hw] = X[tok * SDIM + c];
}

// ---------------------------------------------------------------- host driver
extern "C" void kernel_launch(void* const* d_in, const int* in_sizes, int n_in,
                              void* d_out, int out_size, void* d_ws, size_t ws_size,
                              hipStream_t stream) {
    (void)in_sizes; (void)n_in; (void)out_size; (void)ws_size;
    const float* images  = (const float*)d_in[0];
    const float* pe_w    = (const float*)d_in[1];
    const float* pe_b    = (const float*)d_in[2];
    const float* pe_ln_w = (const float*)d_in[3];
    const float* pe_ln_b = (const float*)d_in[4];

    char* ws = (char*)d_ws;
    size_t off = 0;
    auto alloc = [&](size_t bytes) -> void* {
        void* p = ws + off;
        off += (bytes + 255) & ~(size_t)255;
        return p;
    };
    float* Xa = (float*)alloc((size_t)TOKENS * SDIM * 4);
    float* Xb = (float*)alloc((size_t)TOKENS * SDIM * 4);
    unsigned short* Nbf = (unsigned short*)alloc((size_t)TOKENS * SDIM * 2);
    unsigned short* Big = (unsigned short*)alloc((size_t)TOKENS * 384 * 2);
    unsigned short *Wq[2], *Wp[2], *W1[2], *W2[2];
    for (int bi = 0; bi < 2; bi++) {
        Wq[bi] = (unsigned short*)alloc(27648 * 2);
        Wp[bi] = (unsigned short*)alloc(9216 * 2);
        W1[bi] = (unsigned short*)alloc(36864 * 2);
        W2[bi] = (unsigned short*)alloc(36864 * 2);
    }

    auto cast = [&](const float* s, unsigned short* d, int n) {
        cast_f32_to_bf16<<<(n + 255) / 256, 256, 0, stream>>>(s, d, n);
    };
    cast((const float*)d_in[7],  Wq[0], 27648);
    cast((const float*)d_in[9],  Wp[0], 9216);
    cast((const float*)d_in[14], W1[0], 36864);
    cast((const float*)d_in[16], W2[0], 36864);
    cast((const float*)d_in[20], Wq[1], 27648);
    cast((const float*)d_in[22], Wp[1], 9216);
    cast((const float*)d_in[27], W1[1], 36864);
    cast((const float*)d_in[29], W2[1], 36864);

    long tot = (long)TOKENS * SDIM;
    patch_embed<<<(unsigned)((tot + 255) / 256), 256, 0, stream>>>(images, pe_w, pe_b, Xb);
    layernorm_k<0><<<TOKENS / 8, 256, 0, stream>>>(Xb, pe_ln_w, pe_ln_b, Xa, nullptr);

    for (int bi = 0; bi < 2; bi++) {
        int base = 5 + bi * 13;
        const float* ln1w  = (const float*)d_in[base + 0];
        const float* ln1b  = (const float*)d_in[base + 1];
        const float* qkvb  = (const float*)d_in[base + 3];
        const float* projb = (const float*)d_in[base + 5];
        const float* rpb   = (const float*)d_in[base + 6];
        const float* ln2w  = (const float*)d_in[base + 7];
        const float* ln2b  = (const float*)d_in[base + 8];
        const float* fc1b  = (const float*)d_in[base + 10];
        const float* fc2b  = (const float*)d_in[base + 12];
        int shift = bi ? SSHIFT : 0;

        layernorm_k<1><<<TOKENS / 8, 256, 0, stream>>>(Xa, ln1w, ln1b, nullptr, Nbf);
        gemm_wmma<0, 96, 6><<<dim3(3, TOKENS / 64), 128, 0, stream>>>(
            Nbf, Wq[bi], qkvb, nullptr, nullptr, Big, 288);
        win_attn<<<dim3(4096, 3), 128, 0, stream>>>(Big, rpb, Nbf, shift);
        gemm_wmma<2, 96, 6><<<dim3(1, TOKENS / 64), 128, 0, stream>>>(
            Nbf, Wp[bi], projb, Xa, Xb, nullptr, 96);
        layernorm_k<1><<<TOKENS / 8, 256, 0, stream>>>(Xb, ln2w, ln2b, nullptr, Nbf);
        gemm_wmma<1, 96, 6><<<dim3(4, TOKENS / 64), 128, 0, stream>>>(
            Nbf, W1[bi], fc1b, nullptr, nullptr, Big, 384);
        gemm_wmma<2, 384, 3><<<dim3(2, TOKENS / 64), 128, 0, stream>>>(
            Big, W2[bi], fc2b, Xb, Xa, nullptr, 96);
    }

    to_nchw<<<(unsigned)((tot + 255) / 256), 256, 0, stream>>>(Xa, (float*)d_out);
}
